// Gemma2DecoderLayer_60644938219928
// MI455X (gfx1250) — matrix operand order repaired and checked by reading, not compiled, not run on hardware
//
#include <hip/hip_runtime.h>
#include <math.h>

// ---------------- problem constants ----------------
#define BQ   2
#define SEQ  2048
#define DIM  2304
#define NHQ  8
#define NHKV 4
#define HDIM 256
#define FF   9216
#define WIN  1024

// ---------------- WMMA types (gfx1250, wave32) ----------------
typedef __bf16  bfloat;
typedef bfloat  bf16x16 __attribute__((ext_vector_type(16)));
typedef bfloat  bf16x2  __attribute__((ext_vector_type(2)));
typedef float   floatx8 __attribute__((ext_vector_type(8)));
typedef unsigned int uintx8 __attribute__((ext_vector_type(8)));

// f32 -> bf16 via native converts (backend selects v_cvt*bf16* on gfx1250)
__device__ __forceinline__ unsigned int pack2(float a, float b) {
    bf16x2 h; h[0] = (bfloat)a; h[1] = (bfloat)b;
    return __builtin_bit_cast(unsigned int, h);
}
__device__ __forceinline__ unsigned short bf1(float f) {
    bfloat h = (bfloat)f;
    return __builtin_bit_cast(unsigned short, h);
}
__device__ __forceinline__ bf16x16 make_frag(const uint4 lo, const uint4 hi) {
    uintx8 u;
    u[0] = lo.x; u[1] = lo.y; u[2] = lo.z; u[3] = lo.w;
    u[4] = hi.x; u[5] = hi.y; u[6] = hi.z; u[7] = hi.w;
    return __builtin_bit_cast(bf16x16, u);
}

#define LSTR 40   // LDS row stride in ushort (32 data + 8 pad; 80B keeps b128 alignment)

// =====================================================================
// Generic batched GEMM: C[M,N] = A[M,K] * B(.,.), bf16 WMMA, f32 accum.
//   bNMajor==0 : B is [K,N] row-major (NN)
//   bNMajor==1 : B is [N,K] row-major (NT; i.e. C = A * B^T)
// Batch: grid.z = batchB*batchH; zb=z/batchH, zh=z%batchH.
//   A += zb*sAb + zh*sAh ; B += zb*sBb + (zh/groupB)*sBh ; C += zb*sCb + zh*sCh
// Tiles: 128x128x32, 256 threads (8 waves), each wave: 64x32 accumulator.
// LDS double-buffered: global loads of tile k+1 overlap the WMMAs of tile k.
// =====================================================================
__global__ __launch_bounds__(256)
void gemm_bf16_wmma(const float* __restrict__ A, const float* __restrict__ Bm,
                    float* __restrict__ C,
                    int M, int N, int K, int lda, int ldb, int ldc, int bNMajor,
                    long sAb, long sAh, long sBb, long sBh, long sCb, long sCh,
                    int batchH, int groupB)
{
    __shared__ __align__(16) unsigned short As[2][128 * LSTR];
    __shared__ __align__(16) unsigned short Bs[2][128 * LSTR];

    const int z  = blockIdx.z;
    const int zb = z / batchH;
    const int zh = z % batchH;
    A  += (size_t)zb * sAb + (size_t)zh * sAh;
    Bm += (size_t)zb * sBb + (size_t)(zh / groupB) * sBh;
    C  += (size_t)zb * sCb + (size_t)zh * sCh;

    const int m0 = blockIdx.y * 128;
    const int n0 = blockIdx.x * 128;
    const int t  = threadIdx.x;
    const int lane     = t & 31;
    const int wave     = t >> 5;
    const int waveM    = (wave >> 2) * 64;   // 2 wave rows
    const int waveN    = (wave & 3) * 32;    // 4 wave cols
    const int lane16   = lane & 15;
    const int laneHalf = lane >> 4;

    floatx8 acc[4][2];
    #pragma unroll
    for (int i = 0; i < 4; ++i)
        #pragma unroll
        for (int j = 0; j < 2; ++j)
            #pragma unroll
            for (int e = 0; e < 8; ++e) acc[i][j][e] = 0.f;

    const int arow = t >> 3;        // 0..31 (row within 128, step 32)
    const int acol = (t & 7) * 4;   // 0..28 (k within 32, float4)
    const int bro  = t >> 5;        // 0..7  (k row for NN B tile)
    const int bcol = (t & 31) * 4;  // 0..124 (n col for NN B tile)

    // ---- staged tile in registers ----
    float4 ra[4], rb[4];

    // global load of one 128x32 K-tile into registers
    auto load_tile = [&](int k0) {
        #pragma unroll
        for (int i = 0; i < 4; ++i) {
            const int r = arow + i * 32;
            ra[i] = *(const float4*)(A + (size_t)(m0 + r) * lda + (k0 + acol));
        }
        if (bNMajor) {
            #pragma unroll
            for (int i = 0; i < 4; ++i) {
                const int r = arow + i * 32;    // N row
                rb[i] = *(const float4*)(Bm + (size_t)(n0 + r) * ldb + (k0 + acol));
            }
        } else {
            #pragma unroll
            for (int i = 0; i < 4; ++i) {
                const int kk = bro + i * 8;
                rb[i] = *(const float4*)(Bm + (size_t)(k0 + kk) * ldb + (n0 + bcol));
            }
        }
    };
    // convert + store staged tile into LDS buffer `buf`
    auto store_tile = [&](int buf) {
        #pragma unroll
        for (int i = 0; i < 4; ++i) {
            const int r = arow + i * 32;
            uint2 p; p.x = pack2(ra[i].x, ra[i].y); p.y = pack2(ra[i].z, ra[i].w);
            *(uint2*)&As[buf][r * LSTR + acol] = p;
        }
        if (bNMajor) {
            #pragma unroll
            for (int i = 0; i < 4; ++i) {
                const int r = arow + i * 32;
                uint2 p; p.x = pack2(rb[i].x, rb[i].y); p.y = pack2(rb[i].z, rb[i].w);
                *(uint2*)&Bs[buf][r * LSTR + acol] = p;
            }
        } else {
            #pragma unroll
            for (int i = 0; i < 4; ++i) {
                const int kk = bro + i * 8;
                Bs[buf][(bcol + 0) * LSTR + kk] = bf1(rb[i].x);
                Bs[buf][(bcol + 1) * LSTR + kk] = bf1(rb[i].y);
                Bs[buf][(bcol + 2) * LSTR + kk] = bf1(rb[i].z);
                Bs[buf][(bcol + 3) * LSTR + kk] = bf1(rb[i].w);
            }
        }
    };

    // prologue: tile 0 into buffer 0
    load_tile(0);
    store_tile(0);

    int buf = 0;
    for (int k0 = 0; k0 < K; k0 += 32) {
        __syncthreads();                    // buffer `buf` ready; other buffer free
        const bool hasNext = (k0 + 32 < K);
        if (hasNext) {
            load_tile(k0 + 32);             // VMEM in flight while we do math
            if (k0 + 64 < K) {              // hint tile after next into L2
                __builtin_prefetch(A + (size_t)(m0 + arow) * lda + (k0 + 64 + acol), 0, 1);
                if (bNMajor)
                    __builtin_prefetch(Bm + (size_t)(n0 + arow) * ldb + (k0 + 64 + acol), 0, 1);
                else
                    __builtin_prefetch(Bm + (size_t)(k0 + 64 + bro) * ldb + (n0 + bcol), 0, 1);
            }
        }

        // ---- fragments per ISA layout ----
        // A (16-bit 16x32): lanes 0-15 K=0..7 & 16..23, lanes 16-31 K=8..15 & 24..31
        bf16x16 aF[4];
        #pragma unroll
        for (int i = 0; i < 4; ++i) {
            const int row = waveM + i * 16 + lane16;
            const uint4 lo = *(const uint4*)&As[buf][row * LSTR + laneHalf * 8];
            const uint4 hi = *(const uint4*)&As[buf][row * LSTR + 16 + laneHalf * 8];
            aF[i] = make_frag(lo, hi);
        }
        // B (32x16): lane = column; lanes 0-15 K=0..7 & 16..23, lanes 16-31 K=8..15 & 24..31
        bf16x16 bF[2];
        #pragma unroll
        for (int j = 0; j < 2; ++j) {
            const int col = waveN + j * 16 + lane16;
            const uint4 lo = *(const uint4*)&Bs[buf][col * LSTR + laneHalf * 8];
            const uint4 hi = *(const uint4*)&Bs[buf][col * LSTR + 16 + laneHalf * 8];
            bF[j] = make_frag(lo, hi);
        }
        #pragma unroll
        for (int i = 0; i < 4; ++i)
            #pragma unroll
            for (int j = 0; j < 2; ++j)
                acc[i][j] = __builtin_amdgcn_wmma_f32_16x16x32_bf16(
                    false, aF[i], false, bF[j], (short)0, acc[i][j], false, false);

        if (hasNext) store_tile(buf ^ 1);   // fill the other buffer behind the WMMAs
        buf ^= 1;
    }

    // C/D layout: VGPR r -> M = laneHalf*8 + r ; N = lane16
    #pragma unroll
    for (int i = 0; i < 4; ++i) {
        const int mb = m0 + waveM + i * 16 + laneHalf * 8;
        #pragma unroll
        for (int j = 0; j < 2; ++j) {
            const int col = n0 + waveN + j * 16 + lane16;
            #pragma unroll
            for (int r = 0; r < 8; ++r)
                C[(size_t)(mb + r) * ldc + col] = acc[i][j][r];
        }
    }
}

// =====================================================================
// RMSNorm (optionally fused residual add): out = resid? resid + rms(in) : rms(in)
// =====================================================================
__global__ __launch_bounds__(256)
void rms_kernel(const float* __restrict__ in, const float* __restrict__ scale,
                const float* __restrict__ resid, float* __restrict__ out, int D_)
{
    __shared__ float red[256];
    const size_t row = blockIdx.x;
    const float* x = in + row * D_;
    float ss = 0.f;
    for (int i = threadIdx.x; i < D_; i += 256) { const float v = x[i]; ss += v * v; }
    red[threadIdx.x] = ss;
    __syncthreads();
    for (int s = 128; s > 0; s >>= 1) {
        if (threadIdx.x < s) red[threadIdx.x] += red[threadIdx.x + s];
        __syncthreads();
    }
    const float inv = rsqrtf(red[0] / (float)D_ + 1e-6f);
    float* o = out + row * D_;
    const float* rz = resid ? resid + row * D_ : nullptr;
    for (int i = threadIdx.x; i < D_; i += 256) {
        const float v = x[i] * inv * scale[i];
        o[i] = rz ? rz[i] + v : v;
    }
}

// =====================================================================
// RoPE in place on [B*S, H, 256]; qscale folds the hd^-0.5 for Q.
// =====================================================================
__global__ __launch_bounds__(128)
void rope_kernel(float* __restrict__ x, const int* __restrict__ pos, int H, float qscale)
{
    const int bs = blockIdx.x / H;
    const int h  = blockIdx.x % H;
    const int i  = threadIdx.x;                     // 0..127
    const float p = (float)pos[bs];
    const float freq = powf(10000.f, -(float)i / 128.f);
    float s, c;
    sincosf(p * freq, &s, &c);
    float* base = x + ((size_t)bs * H + h) * HDIM;
    const float x1 = base[i] * qscale;
    const float x2 = base[i + 128] * qscale;
    base[i]        = x1 * c - x2 * s;
    base[i + 128]  = x2 * c + x1 * s;
}

// =====================================================================
// Soft-cap + causal/window/segment mask + softmax, one block per logit row.
// =====================================================================
__global__ __launch_bounds__(256)
void softmax_cap_mask(float* __restrict__ logits, const int* __restrict__ pos,
                      const int* __restrict__ seg, int window)
{
    const int q = blockIdx.x, h = blockIdx.y, b = blockIdx.z;
    float* row = logits + (((size_t)b * NHQ + h) * SEQ + q) * SEQ;
    const int pq = pos[b * SEQ + q];
    const int sq = seg[b * SEQ + q];
    __shared__ float red[256];
    float lv[SEQ / 256];
    float lmax = -3.0e38f;
    #pragma unroll
    for (int it = 0; it < SEQ / 256; ++it) {
        const int s  = threadIdx.x + it * 256;
        const int ps = pos[b * SEQ + s];
        float l = 50.f * tanhf(row[s] * 0.02f);
        bool ok = (pq >= ps) && (sq == seg[b * SEQ + s]);
        if (window > 0) ok = ok && ((pq - ps) < window);
        l = ok ? l : -1e30f;
        lv[it] = l;
        lmax = fmaxf(lmax, l);
    }
    red[threadIdx.x] = lmax;
    __syncthreads();
    for (int s2 = 128; s2 > 0; s2 >>= 1) {
        if (threadIdx.x < s2) red[threadIdx.x] = fmaxf(red[threadIdx.x], red[threadIdx.x + s2]);
        __syncthreads();
    }
    lmax = red[0];
    __syncthreads();
    float sum = 0.f;
    #pragma unroll
    for (int it = 0; it < SEQ / 256; ++it) { lv[it] = __expf(lv[it] - lmax); sum += lv[it]; }
    red[threadIdx.x] = sum;
    __syncthreads();
    for (int s2 = 128; s2 > 0; s2 >>= 1) {
        if (threadIdx.x < s2) red[threadIdx.x] += red[threadIdx.x + s2];
        __syncthreads();
    }
    const float rinv = 1.f / red[0];
    #pragma unroll
    for (int it = 0; it < SEQ / 256; ++it)
        row[threadIdx.x + it * 256] = lv[it] * rinv;
}

// =====================================================================
// g0 = gelu_tanh(g0) * g1
// =====================================================================
__global__ __launch_bounds__(256)
void gelu_mul(float* __restrict__ g0, const float* __restrict__ g1, size_t n)
{
    for (size_t i = (size_t)blockIdx.x * 256 + threadIdx.x; i < n;
         i += (size_t)gridDim.x * 256) {
        const float x = g0[i];
        const float t = 0.7978845608028654f * (x + 0.044715f * x * x * x);
        g0[i] = 0.5f * x * (1.f + tanhf(t)) * g1[i];
    }
}

// ---------------- host-side orchestration ----------------
struct BlockW {
    const float *pre_attn, *post_attn, *pre_ffw, *post_ffw;
    const float *wq, *wk, *wv, *wo, *wi0, *wi1, *wom;
};

static inline void launch_gemm(hipStream_t st, const float* A, const float* B, float* C,
                               int M, int N, int K, int lda, int ldb, int ldc, int bNMajor,
                               long sAb = 0, long sAh = 0, long sBb = 0, long sBh = 0,
                               long sCb = 0, long sCh = 0,
                               int batchB = 1, int batchH = 1, int groupB = 1)
{
    dim3 grid(N / 128, M / 128, batchB * batchH);
    gemm_bf16_wmma<<<grid, 256, 0, st>>>(A, B, C, M, N, K, lda, ldb, ldc, bNMajor,
                                         sAb, sAh, sBb, sBh, sCb, sCh, batchH, groupB);
}

static void run_block(hipStream_t st, float* XC, float* H, float* Q, float* Kb, float* Vb,
                      float* O, float* Ab, float* G0, float* G1, float* SC,
                      const int* pos, const int* seg, const BlockW& w, int window, float* outx)
{
    const int M = BQ * SEQ;                               // 4096
    // ---- attention ----
    rms_kernel<<<M, 256, 0, st>>>(XC, w.pre_attn, nullptr, H, DIM);
    launch_gemm(st, H, w.wq, Q,  M, NHQ  * HDIM, DIM, DIM, NHQ  * HDIM, NHQ  * HDIM, 0);
    launch_gemm(st, H, w.wk, Kb, M, NHKV * HDIM, DIM, DIM, NHKV * HDIM, NHKV * HDIM, 0);
    launch_gemm(st, H, w.wv, Vb, M, NHKV * HDIM, DIM, DIM, NHKV * HDIM, NHKV * HDIM, 0);
    rope_kernel<<<M * NHQ, 128, 0, st>>>(Q, pos, NHQ, 0.0625f);   // hd^-0.5 folded in
    rope_kernel<<<M * NHKV, 128, 0, st>>>(Kb, pos, NHKV, 1.f);
    // scores[b,h,q,s] = q . k  (NT, batched over B*HQ, GQA group 2)
    launch_gemm(st, Q, Kb, SC, SEQ, SEQ, HDIM,
                NHQ * HDIM, NHKV * HDIM, SEQ, 1,
                (long)SEQ * NHQ * HDIM, (long)HDIM,
                (long)SEQ * NHKV * HDIM, (long)HDIM,
                (long)NHQ * SEQ * SEQ, (long)SEQ * SEQ,
                BQ, NHQ, NHQ / NHKV);
    softmax_cap_mask<<<dim3(SEQ, NHQ, BQ), 256, 0, st>>>(SC, pos, seg, window);
    // o[b,q,h,:] = p . v  (NN, batched; C strided into [B,S,HQ,HD])
    launch_gemm(st, SC, Vb, O, SEQ, HDIM, SEQ,
                SEQ, NHKV * HDIM, NHQ * HDIM, 0,
                (long)NHQ * SEQ * SEQ, (long)SEQ * SEQ,
                (long)SEQ * NHKV * HDIM, (long)HDIM,
                (long)SEQ * NHQ * HDIM, (long)HDIM,
                BQ, NHQ, NHQ / NHKV);
    launch_gemm(st, O, w.wo, Ab, M, DIM, NHQ * HDIM, NHQ * HDIM, DIM, DIM, 0);
    rms_kernel<<<M, 256, 0, st>>>(Ab, w.post_attn, XC, XC, DIM);  // x += rms(attn)
    // ---- FFN ----
    rms_kernel<<<M, 256, 0, st>>>(XC, w.pre_ffw, nullptr, H, DIM);
    launch_gemm(st, H, w.wi0, G0, M, FF, DIM, DIM, FF, FF, 0);
    launch_gemm(st, H, w.wi1, G1, M, FF, DIM, DIM, FF, FF, 0);
    gelu_mul<<<4096, 256, 0, st>>>(G0, G1, (size_t)M * FF);
    launch_gemm(st, G0, w.wom, Ab, M, DIM, FF, FF, DIM, DIM, 0);
    rms_kernel<<<M, 256, 0, st>>>(Ab, w.post_ffw, XC, outx, DIM); // out = x + rms(ffn)
}

extern "C" void kernel_launch(void* const* d_in, const int* in_sizes, int n_in,
                              void* d_out, int out_size, void* d_ws, size_t ws_size,
                              hipStream_t stream)
{
    (void)in_sizes; (void)n_in; (void)out_size; (void)ws_size;
    const float* x  = (const float*)d_in[0];
    const int* pos  = (const int*)d_in[1];
    const int* seg  = (const int*)d_in[2];
    const BlockW wl = { (const float*)d_in[3],  (const float*)d_in[4],
                        (const float*)d_in[5],  (const float*)d_in[6],
                        (const float*)d_in[7],  (const float*)d_in[8],
                        (const float*)d_in[9],  (const float*)d_in[10],
                        (const float*)d_in[11], (const float*)d_in[12],
                        (const float*)d_in[13] };
    const BlockW wg = { (const float*)d_in[14], (const float*)d_in[15],
                        (const float*)d_in[16], (const float*)d_in[17],
                        (const float*)d_in[18], (const float*)d_in[19],
                        (const float*)d_in[20], (const float*)d_in[21],
                        (const float*)d_in[22], (const float*)d_in[23],
                        (const float*)d_in[24] };

    float* ws = (float*)d_ws;
    size_t off = 0;
    float* XC = ws + off; off += (size_t)BQ * SEQ * DIM;          // running residual x
    float* H  = ws + off; off += (size_t)BQ * SEQ * DIM;          // rms output
    float* Q  = ws + off; off += (size_t)BQ * SEQ * NHQ  * HDIM;
    float* Kb = ws + off; off += (size_t)BQ * SEQ * NHKV * HDIM;
    float* Vb = ws + off; off += (size_t)BQ * SEQ * NHKV * HDIM;
    float* O  = ws + off; off += (size_t)BQ * SEQ * NHQ  * HDIM;
    float* Ab = ws + off; off += (size_t)BQ * SEQ * DIM;          // proj out / ffn out
    float* G0 = ws + off; off += (size_t)BQ * SEQ * FF;
    float* G1 = ws + off; off += (size_t)BQ * SEQ * FF;
    float* SC = G0;   // scores (B*HQ*S*S = 268MB) alias FFN scratch (302MB), disjoint in time

    hipMemcpyAsync(XC, x, sizeof(float) * (size_t)BQ * SEQ * DIM,
                   hipMemcpyDeviceToDevice, stream);

    run_block(stream, XC, H, Q, Kb, Vb, O, Ab, G0, G1, SC, pos, seg, wl, WIN, XC);
    run_block(stream, XC, H, Q, Kb, Vb, O, Ab, G0, G1, SC, pos, seg, wg, 0, (float*)d_out);
}
